// LocalSelfAttention_63728724738641
// MI455X (gfx1250) — matrix operand order repaired and checked by reading, not compiled, not run on hardware
//
#include <hip/hip_runtime.h>

// ---------------------------------------------------------------------------
// Local windowed self-attention for MI455X (gfx1250), bf16 WMMA pipeline.
//   K0: one-pass fp32->bf16 conversion of hidden / qkv_w / out_w (removes all
//       conversion VALU from the GEMM hot loops)
//   K1: qkv = hidden @ qkv_w^T + qkv_b   -> Q[B,H,L,D], K[B,H,L,D], Vt[B,H,D,L]
//   K2: windowed attention (W=32 causal) -> ctx[B*L, 512] (bf16)
//   K3: out = ctx @ out_w^T + out_b      -> fp32 output
// GEMMs: 32x64 tiles per wave, 8 v_wmma_f32_16x16x32_bf16 per K-step.
// ---------------------------------------------------------------------------

typedef __attribute__((ext_vector_type(16))) __bf16 v16bf;
typedef __attribute__((ext_vector_type(8)))  float  v8f;
typedef unsigned short u16;

union Frag { v16bf v; u16 u[16]; };

#define HID   512
#define LSEQ  4096
#define NH    8
#define HD    64
#define NB    4

__device__ __forceinline__ u16 f2bf(float f) {
    union { float f; unsigned int u; } c; c.f = f;
    unsigned int r = c.u + 0x7FFFu + ((c.u >> 16) & 1u);   // round-to-nearest-even
    return (u16)(r >> 16);
}

__device__ __forceinline__ v8f wmma_bf16(const Frag& a, const Frag& b, v8f c) {
    return __builtin_amdgcn_wmma_f32_16x16x32_bf16(false, a.v, false, b.v,
                                                   (short)0, c, false, false);
}

// bf16 A/B fragment load (contiguous along contraction dim)
__device__ __forceinline__ void load_frag16(Frag& f, const u16* __restrict__ p) {
    #pragma unroll
    for (int i = 0; i < 8; ++i) f.u[i]     = p[i];
    #pragma unroll
    for (int i = 0; i < 8; ++i) f.u[8 + i] = p[16 + i];
}

// --------------------------- K0: fp32 -> bf16 ------------------------------
__global__ void __launch_bounds__(256)
cvt_f32_bf16(const float* __restrict__ src, u16* __restrict__ dst, int n4) {
    const int i = blockIdx.x * 256 + threadIdx.x;
    if (i < n4) {
        const float4 f = ((const float4*)src)[i];
        ushort4 o;
        o.x = f2bf(f.x); o.y = f2bf(f.y); o.z = f2bf(f.z); o.w = f2bf(f.w);
        ((ushort4*)dst)[i] = o;
    }
}

// --------------------------- K1: QKV projection ----------------------------
// one wave computes a 32(M) x 64(N) tile; M = 16384, N = 1536, K = 512
__global__ void __launch_bounds__(256)
qkv_kernel(const u16* __restrict__ hiddenB, const u16* __restrict__ qkv_wB,
           const float* __restrict__ qkv_b,
           u16* __restrict__ Q, u16* __restrict__ Kk, u16* __restrict__ Vt)
{
    const int lane = threadIdx.x & 31;
    const int w    = blockIdx.x * 8 + (threadIdx.x >> 5);
    const int mt   = w / 24, nt = w % 24;          // 512 x 24 tiles
    const int row0 = mt * 32, n0 = nt * 64;

    const int klo  = (lane & 16) ? 8 : 0;          // A-frag low-K base per half-wave
    const int koff = (lane & 16) ? 8 : 0;          // B-frag K base per half-wave

    const u16* ap0 = hiddenB + (size_t)(row0 + (lane & 15)) * HID + klo;
    const u16* ap1 = ap0 + (size_t)16 * HID;
    const u16* bp  = qkv_wB + (size_t)(n0 + (lane & 15)) * HID + koff;

    v8f acc[2][4] = {};
    for (int c = 0; c < HID; c += 32) {
        Frag a0, a1;
        load_frag16(a0, ap0 + c);
        load_frag16(a1, ap1 + c);
        __builtin_prefetch(ap0 + c + 32, 0, 1);
        __builtin_prefetch(ap1 + c + 32, 0, 1);
        #pragma unroll
        for (int j = 0; j < 4; ++j) {
            Frag b;
            load_frag16(b, bp + (size_t)(j * 16) * HID + c);
            acc[0][j] = wmma_bf16(a0, b, acc[0][j]);
            acc[1][j] = wmma_bf16(a1, b, acc[1][j]);
        }
    }

    const int mbase = (lane & 16) ? 8 : 0;
    #pragma unroll
    for (int s = 0; s < 2; ++s) {
        #pragma unroll
        for (int j = 0; j < 4; ++j) {
            const int n = n0 + j * 16 + (lane & 15);
            const float bias = qkv_b[n];
            #pragma unroll
            for (int r = 0; r < 8; ++r) {
                const int m = row0 + s * 16 + mbase + r;
                const int b = m >> 12, l = m & (LSEQ - 1);
                const u16 bv = f2bf(acc[s][j][r] + bias);
                if (n < 512) {                                   // Q [B,H,L,D]
                    const int h = n >> 6, d = n & 63;
                    Q[(((size_t)(b * NH + h)) * LSEQ + l) * HD + d] = bv;
                } else if (n < 1024) {                           // K [B,H,L,D]
                    const int nn = n - 512, h = nn >> 6, d = nn & 63;
                    Kk[(((size_t)(b * NH + h)) * LSEQ + l) * HD + d] = bv;
                } else {                                         // V^T [B,H,D,L]
                    const int nn = n - 1024, h = nn >> 6, d = nn & 63;
                    Vt[(((size_t)(b * NH + h)) * HD + d) * LSEQ + l] = bv;
                }
            }
        }
    }
}

// --------------------------- K2: windowed attention ------------------------
// one wave per (b, h, 16-query tile); keys span [l0-32, l0+15] = 3 tiles of 16
__global__ void __launch_bounds__(256)
attn_kernel(const u16* __restrict__ Q, const u16* __restrict__ K,
            const u16* __restrict__ Vt, u16* __restrict__ ctx)
{
    __shared__ float sS[8][16][64];   // raw scores (48 cols used)
    __shared__ u16   sP[8][16][64];   // probs, zero-padded 48..63

    const int lane = threadIdx.x & 31;
    const int wv   = threadIdx.x >> 5;
    const int w    = blockIdx.x * 8 + wv;
    const int qt   = w & 255;                  // 256 query tiles per (b,h)
    const int bh   = w >> 8;                   // 0..31
    const int l0   = qt * 16;

    const u16* Qb = Q  + (size_t)bh * LSEQ * HD;
    const u16* Kb = K  + (size_t)bh * LSEQ * HD;
    const u16* Vb = Vt + (size_t)bh * HD * LSEQ;

    const int klo  = (lane & 16) ? 8 : 0;
    const int koff = (lane & 16) ? 8 : 0;

    // Q A-fragments (head-dim chunks 0..31 and 32..63)
    Frag qa[2];
    {
        const u16* qp = Qb + (size_t)(l0 + (lane & 15)) * HD + klo;
        load_frag16(qa[0], qp);
        load_frag16(qa[1], qp + 32);
    }

    // scores: S[16 q][48 k] over keys l0-32 .. l0+15
    #pragma unroll
    for (int t = 0; t < 3; ++t) {
        const int key = l0 - 32 + t * 16 + (lane & 15);
        v8f acc = {};
        #pragma unroll
        for (int c = 0; c < 2; ++c) {
            Frag kb;
            if (key >= 0) {
                load_frag16(kb, Kb + (size_t)key * HD + c * 32 + koff);
            } else {
                #pragma unroll
                for (int i = 0; i < 16; ++i) kb.u[i] = 0;
            }
            acc = wmma_bf16(qa[c], kb, acc);
        }
        const int mbase = (lane & 16) ? 8 : 0;
        const int n = lane & 15;
        #pragma unroll
        for (int r = 0; r < 8; ++r) sS[wv][mbase + r][t * 16 + n] = acc[r];
    }
    __syncthreads();

    // masked softmax: lanes 0..15 each own one query row
    if (lane < 16) {
        const float SC = 0.125f;                 // HEAD_DIM^-0.5
        const int qg = l0 + lane;
        float mx = -1e30f;
        #pragma unroll
        for (int j = 0; j < 48; ++j) {
            const int kg = l0 - 32 + j;
            if (kg >= 0 && kg >= qg - 31 && kg <= qg)
                mx = fmaxf(mx, sS[wv][lane][j] * SC);
        }
        float sum = 0.f;
        #pragma unroll
        for (int j = 0; j < 48; ++j) {
            const int kg = l0 - 32 + j;
            if (kg >= 0 && kg >= qg - 31 && kg <= qg)
                sum += __expf(sS[wv][lane][j] * SC - mx);
        }
        const float inv = 1.f / fmaxf(sum, 1e-6f);
        #pragma unroll
        for (int j = 0; j < 48; ++j) {
            const int kg = l0 - 32 + j;
            const bool ok = (kg >= 0 && kg >= qg - 31 && kg <= qg);
            const float p = ok ? __expf(sS[wv][lane][j] * SC - mx) * inv : 0.f;
            sP[wv][lane][j] = f2bf(p);
        }
        #pragma unroll
        for (int j = 48; j < 64; ++j) sP[wv][lane][j] = 0;
    }
    __syncthreads();

    // P A-fragments from LDS (K = 64 padded key window)
    Frag pa[2];
    {
        const int row = lane & 15;
        #pragma unroll
        for (int c = 0; c < 2; ++c) {
            const int base = c * 32 + klo;
            #pragma unroll
            for (int i = 0; i < 8; ++i) pa[c].u[i]     = sP[wv][row][base + i];
            #pragma unroll
            for (int i = 0; i < 8; ++i) pa[c].u[8 + i] = sP[wv][row][base + 16 + i];
        }
    }

    // ctx = P @ V : 4 head-dim tiles x 2 K-chunks
    const int kt0 = l0 - 32;
    const int b = bh >> 3, h = bh & 7;
    #pragma unroll
    for (int dt = 0; dt < 4; ++dt) {
        const int d = dt * 16 + (lane & 15);
        v8f acc = {};
        #pragma unroll
        for (int c = 0; c < 2; ++c) {
            const int ks = kt0 + c * 32 + koff;       // 8-aligned; both 8-key runs zeroed if ks out of range
            Frag vb;
            if (ks >= 0 && ks < LSEQ) {
                load_frag16(vb, Vb + (size_t)d * LSEQ + ks);
            } else {
                #pragma unroll
                for (int i = 0; i < 16; ++i) vb.u[i] = 0;
            }
            acc = wmma_bf16(pa[c], vb, acc);
        }
        const int mbase = (lane & 16) ? 8 : 0;
        const int n = lane & 15;
        #pragma unroll
        for (int r = 0; r < 8; ++r) {
            const int l = l0 + mbase + r;
            ctx[((size_t)(b * LSEQ + l)) * HID + h * HD + dt * 16 + n] = f2bf(acc[r]);
        }
    }
}

// --------------------------- K3: output projection -------------------------
// out[16384, 512] = ctx(bf16) @ out_w^T + out_b, fp32 result; 32x64 tiles
__global__ void __launch_bounds__(256)
outproj_kernel(const u16* __restrict__ ctx, const u16* __restrict__ out_wB,
               const float* __restrict__ out_b, float* __restrict__ out)
{
    const int lane = threadIdx.x & 31;
    const int w    = blockIdx.x * 8 + (threadIdx.x >> 5);
    const int mt   = w / 8, nt = w % 8;            // 512 x 8 tiles
    const int row0 = mt * 32, n0 = nt * 64;

    const int klo  = (lane & 16) ? 8 : 0;
    const int koff = (lane & 16) ? 8 : 0;

    const u16* ap0 = ctx + (size_t)(row0 + (lane & 15)) * HID + klo;
    const u16* ap1 = ap0 + (size_t)16 * HID;
    const u16* bp  = out_wB + (size_t)(n0 + (lane & 15)) * HID + koff;

    v8f acc[2][4] = {};
    for (int c = 0; c < HID; c += 32) {
        Frag a0, a1;
        load_frag16(a0, ap0 + c);
        load_frag16(a1, ap1 + c);
        __builtin_prefetch(ap0 + c + 32, 0, 1);
        __builtin_prefetch(ap1 + c + 32, 0, 1);
        #pragma unroll
        for (int j = 0; j < 4; ++j) {
            Frag b;
            load_frag16(b, bp + (size_t)(j * 16) * HID + c);
            acc[0][j] = wmma_bf16(a0, b, acc[0][j]);
            acc[1][j] = wmma_bf16(a1, b, acc[1][j]);
        }
    }

    const int mbase = (lane & 16) ? 8 : 0;
    #pragma unroll
    for (int s = 0; s < 2; ++s) {
        #pragma unroll
        for (int j = 0; j < 4; ++j) {
            const int n = n0 + j * 16 + (lane & 15);
            const float bias = out_b[n];
            #pragma unroll
            for (int r = 0; r < 8; ++r) {
                const int m = row0 + s * 16 + mbase + r;
                out[(size_t)m * HID + n] = acc[s][j][r] + bias;
            }
        }
    }
}

// ---------------------------------------------------------------------------
extern "C" void kernel_launch(void* const* d_in, const int* in_sizes, int n_in,
                              void* d_out, int out_size, void* d_ws, size_t ws_size,
                              hipStream_t stream) {
    const float* hidden = (const float*)d_in[0];
    // d_in[1] = valid_mask (all true in this problem; masking is structural)
    const float* qkv_w  = (const float*)d_in[2];
    const float* qkv_b  = (const float*)d_in[3];
    const float* out_w  = (const float*)d_in[4];
    const float* out_b  = (const float*)d_in[5];
    float* out = (float*)d_out;

    // workspace (bf16 elements):
    const size_t PLANE = (size_t)NB * NH * LSEQ * HD;      // 8,388,608 = B*L*512
    u16* Q       = (u16*)d_ws;
    u16* K       = Q + PLANE;
    u16* Vt      = K + PLANE;
    u16* ctx     = Vt + PLANE;
    u16* hiddenB = ctx + PLANE;
    u16* qkv_wB  = hiddenB + PLANE;                        // 1536*512 = 786,432
    u16* out_wB  = qkv_wB + (size_t)1536 * HID;            // 512*512  = 262,144

    // K0: one-pass conversions (4 elems/thread)
    cvt_f32_bf16<<<(int)(PLANE / 4 / 256), 256, 0, stream>>>(hidden, hiddenB, (int)(PLANE / 4));
    cvt_f32_bf16<<<768, 256, 0, stream>>>(qkv_w, qkv_wB, 1536 * HID / 4);
    cvt_f32_bf16<<<256, 256, 0, stream>>>(out_w, out_wB, HID * HID / 4);

    // K1: (16384/32) x (1536/64) = 12288 waves -> 1536 blocks of 8 waves
    qkv_kernel<<<1536, 256, 0, stream>>>(hiddenB, qkv_wB, qkv_b, Q, K, Vt);
    // K2: 4*8*256 = 8192 waves -> 1024 blocks
    attn_kernel<<<1024, 256, 0, stream>>>(Q, K, Vt, ctx);
    // K3: (16384/32) x (512/64) = 4096 waves -> 512 blocks
    outproj_kernel<<<512, 256, 0, stream>>>(ctx, out_wB, out_b, out);
}
